// OwnGCN_73443940761885
// MI455X (gfx1250) — matrix operand order repaired and checked by reading, not compiled, not run on hardware
//
#include <hip/hip_runtime.h>

// ---------------------------------------------------------------------------
// Types for CDNA5 WMMA
// ---------------------------------------------------------------------------
typedef __attribute__((ext_vector_type(16))) __bf16 v16bf;
typedef __attribute__((ext_vector_type(8)))  float  v8f;

#define N_LAYERS 16

__device__ __forceinline__ unsigned short f2bf(float f) {
    unsigned int u = __float_as_uint(f);
    unsigned int r = (u + 0x7FFFu + ((u >> 16) & 1u)) >> 16;   // round-to-nearest-even
    return (unsigned short)r;
}

__device__ __forceinline__ float bf2f(unsigned short h) {
    return __uint_as_float((unsigned int)h << 16);
}

// ---------------------------------------------------------------------------
// Elementwise / utility kernels
// ---------------------------------------------------------------------------
__global__ void k_zero_f32(float* __restrict__ p, size_t n) {
    size_t i = (size_t)blockIdx.x * blockDim.x + threadIdx.x;
    size_t stride = (size_t)gridDim.x * blockDim.x;
    for (; i < n; i += stride) p[i] = 0.0f;
}

__global__ void k_copy_f32(const float* __restrict__ src, float* __restrict__ dst, size_t n) {
    size_t i = (size_t)blockIdx.x * blockDim.x + threadIdx.x;
    size_t stride = (size_t)gridDim.x * blockDim.x;
    for (; i < n; i += stride) dst[i] = src[i];
}

// dst = -src   (seed buffer for T2 = 2*prop(T1) - T0)
__global__ void k_negcopy_f32(const float* __restrict__ src, float* __restrict__ dst, size_t n) {
    size_t i = (size_t)blockIdx.x * blockDim.x + threadIdx.x;
    size_t stride = (size_t)gridDim.x * blockDim.x;
    for (; i < n; i += stride) dst[i] = -src[i];
}

__global__ void k_f32_to_bf16(const float* __restrict__ in, unsigned short* __restrict__ out, size_t n) {
    size_t i = (size_t)blockIdx.x * blockDim.x + threadIdx.x;
    size_t stride = (size_t)gridDim.x * blockDim.x;
    for (; i < n; i += stride) out[i] = f2bf(in[i]);
}

// Wt[n*cin + k] = bf16( W[k*cout + n] )   (transpose + convert)
__global__ void k_w_bf16_t(const float* __restrict__ W, unsigned short* __restrict__ Wt,
                           int cin, int cout) {
    size_t total = (size_t)cin * cout;
    size_t i = (size_t)blockIdx.x * blockDim.x + threadIdx.x;
    size_t stride = (size_t)gridDim.x * blockDim.x;
    for (; i < total; i += stride) {
        int n = (int)(i / cin);
        int k = (int)(i % cin);
        Wt[i] = f2bf(W[(size_t)k * cout + n]);
    }
}

// out = relu(acc + x)
__global__ void k_resid_relu(const float* __restrict__ acc, const float* __restrict__ x,
                             float* __restrict__ out, size_t n) {
    size_t i = (size_t)blockIdx.x * blockDim.x + threadIdx.x;
    size_t stride = (size_t)gridDim.x * blockDim.x;
    for (; i < n; i += stride) {
        float v = acc[i] + x[i];
        out[i] = v > 0.0f ? v : 0.0f;
    }
}

// ---------------------------------------------------------------------------
// Graph structure kernels
// ---------------------------------------------------------------------------
__global__ void k_degree(const int* __restrict__ src, const int* __restrict__ dst,
                         float* __restrict__ deg, int E) {
    int i = blockIdx.x * blockDim.x + threadIdx.x;
    int stride = gridDim.x * blockDim.x;
    for (; i < E; i += stride) {
        if (src[i] != dst[i]) atomicAdd(&deg[src[i]], 1.0f);
    }
}

__global__ void k_dis(const float* __restrict__ deg, float* __restrict__ dis, int N) {
    int i = blockIdx.x * blockDim.x + threadIdx.x;
    int stride = gridDim.x * blockDim.x;
    for (; i < N; i += stride) {
        float d = deg[i];
        dis[i] = (d > 0.0f) ? rsqrtf(fmaxf(d, 1.0f)) : 0.0f;
    }
}

__global__ void k_edge_norm(const int* __restrict__ src, const int* __restrict__ dst,
                            const float* __restrict__ dis, float* __restrict__ nrm, int E) {
    int i = blockIdx.x * blockDim.x + threadIdx.x;
    int stride = gridDim.x * blockDim.x;
    for (; i < E; i += stride) {
        float w = (src[i] != dst[i]) ? 1.0f : 0.0f;
        nrm[i] = -dis[src[i]] * w * dis[dst[i]];
    }
}

// Out[dst,:] += scale * norm[e] * bf16(H[src,:])
// Gather side reads the bf16 staging copy of the Chebyshev term (it already
// exists for the WMMA GEMM), halving the random-gather HBM traffic; the
// segment-sum accumulation stays fp32 via global_atomic_add_f32.
__global__ void k_spmm_scatter(const int* __restrict__ src, const int* __restrict__ dst,
                               const float* __restrict__ nrm,
                               const unsigned short* __restrict__ Hbf,
                               float* __restrict__ Out, int C, int E, float scale) {
    for (int e = blockIdx.x; e < E; e += gridDim.x) {
        float w = nrm[e] * scale;
        if (w == 0.0f) continue;
        const unsigned short* hr = Hbf + (size_t)src[e] * C;
        float* orow = Out + (size_t)dst[e] * C;
        for (int c = threadIdx.x; c < C; c += blockDim.x)
            atomicAdd(&orow[c], w * bf2f(hr[c]));
    }
}

// ---------------------------------------------------------------------------
// GraphNorm kernels (single-pass column stats: mean + shifted variance)
// var = E[h^2] - 2*s*E[h] + s^2, with s = mean*mean_scale
// ---------------------------------------------------------------------------
__global__ void k_col_stats(const float* __restrict__ H, const float* __restrict__ ms,
                            float* __restrict__ mean, float* __restrict__ var,
                            int Nrows, int C) {
    __shared__ float rs[256];
    __shared__ float rq[256];
    int c = blockIdx.x;
    float s = 0.0f, q = 0.0f;
    for (int r = threadIdx.x; r < Nrows; r += blockDim.x) {
        float v = H[(size_t)r * C + c];
        s += v;
        q += v * v;
    }
    rs[threadIdx.x] = s;
    rq[threadIdx.x] = q;
    __syncthreads();
    for (int off = blockDim.x >> 1; off > 0; off >>= 1) {
        if (threadIdx.x < off) {
            rs[threadIdx.x] += rs[threadIdx.x + off];
            rq[threadIdx.x] += rq[threadIdx.x + off];
        }
        __syncthreads();
    }
    if (threadIdx.x == 0) {
        float m   = rs[0] / (float)Nrows;
        float msq = rq[0] / (float)Nrows;
        float sh  = m * ms[c];
        mean[c] = m;
        var[c]  = msq - 2.0f * sh * m + sh * sh;
    }
}

__global__ void k_gn_leaky(const float* __restrict__ Hin, float* __restrict__ Hout,
                           const float* __restrict__ mean, const float* __restrict__ var,
                           const float* __restrict__ gw, const float* __restrict__ gb,
                           const float* __restrict__ ms, size_t total, int C) {
    size_t i = (size_t)blockIdx.x * blockDim.x + threadIdx.x;
    size_t stride = (size_t)gridDim.x * blockDim.x;
    for (; i < total; i += stride) {
        int c = (int)(i % C);
        float o = Hin[i] - mean[c] * ms[c];
        float v = gw[c] * o * rsqrtf(var[c] + 1e-5f) + gb[c];
        Hout[i] = v > 0.0f ? v : 0.2f * v;
    }
}

// Column mean only (for the global mean pool)
__global__ void k_col_mean(const float* __restrict__ H, float* __restrict__ mean,
                           int Nrows, int C) {
    __shared__ float red[256];
    int c = blockIdx.x;
    float s = 0.0f;
    for (int r = threadIdx.x; r < Nrows; r += blockDim.x)
        s += H[(size_t)r * C + c];
    red[threadIdx.x] = s;
    __syncthreads();
    for (int off = blockDim.x >> 1; off > 0; off >>= 1) {
        if (threadIdx.x < off) red[threadIdx.x] += red[threadIdx.x + off];
        __syncthreads();
    }
    if (threadIdx.x == 0) mean[c] = red[0] / (float)Nrows;
}

// ---------------------------------------------------------------------------
// Fused multi-term WMMA GEMM:
//   Out[N x Ncols] = bias + sum_t  A_t[N x K] @ Bt_t[Ncols x K]^T     (t < NTERMS)
// Accumulator lives entirely in the v8f fragments (no C read/write round-trips).
// Each wave computes a 16x32 output tile (2 N-tiles): one A fragment feeds two
// v_wmma_f32_16x16x32_bf16, halving A-side loads per wmma.
// Fragment packing: contiguous k order (lane half h holds k 16h..16h+15), same for A and B.
// ---------------------------------------------------------------------------
template <int NTERMS>
__global__ void k_wmma_gemm_fused(const unsigned short* __restrict__ A0,
                                  const unsigned short* __restrict__ A1,
                                  const unsigned short* __restrict__ A2,
                                  const unsigned short* __restrict__ B0,
                                  const unsigned short* __restrict__ B1,
                                  const unsigned short* __restrict__ B2,
                                  const float* __restrict__ bias,
                                  float* __restrict__ Out,
                                  int Nrows, int Kdim, int Ncols) {
    const int lane = threadIdx.x & 31;
    const int wave = threadIdx.x >> 5;
    const int mtile = blockIdx.x * (blockDim.x >> 5) + wave;
    const int nt0 = blockIdx.y * 2;          // two adjacent 16-col tiles per wave
    if (mtile * 16 >= Nrows) return;

    const int halfsel = lane >> 4;           // 0: lanes 0-15, 1: lanes 16-31
    const int l16 = lane & 15;

    // Accumulators seeded with bias (bias depends on column only)
    float bv0 = bias[nt0 * 16 + l16];
    float bv1 = bias[nt0 * 16 + 16 + l16];
    v8f c0, c1;
    #pragma unroll
    for (int i = 0; i < 8; ++i) { c0[i] = bv0; c1[i] = bv1; }

    #pragma unroll
    for (int t = 0; t < NTERMS; ++t) {
        const unsigned short* A  = (t == 0) ? A0 : (t == 1) ? A1 : A2;
        const unsigned short* Bt = (t == 0) ? B0 : (t == 1) ? B1 : B2;
        const unsigned short* Arow = A  + (size_t)(mtile * 16 + l16) * Kdim;
        const unsigned short* Bc0  = Bt + (size_t)(nt0 * 16 + l16) * Kdim;
        const unsigned short* Bc1  = Bt + (size_t)(nt0 * 16 + 16 + l16) * Kdim;

        for (int k0 = 0; k0 < Kdim; k0 += 32) {
            union Frag { v16bf v; float4 f[2]; };
            Frag a, b0, b1;
            // A 16-bit 16x32: lanes 0-15 hold K 0..15; lanes 16-31 hold K 16..31
            const float4* ap = (const float4*)(Arow + k0);
            a.f[0] = ap[halfsel * 2];
            a.f[1] = ap[halfsel * 2 + 1];
            // B 16-bit 32x16: lanes 0-15 hold K 0..15 of column l16; lanes 16-31 hold K 16..31
            const float4* bp0 = (const float4*)(Bc0 + k0);
            b0.f[0] = bp0[halfsel * 2];
            b0.f[1] = bp0[halfsel * 2 + 1];
            const float4* bp1 = (const float4*)(Bc1 + k0);
            b1.f[0] = bp1[halfsel * 2];
            b1.f[1] = bp1[halfsel * 2 + 1];
            c0 = __builtin_amdgcn_wmma_f32_16x16x32_bf16(false, a.v, false, b0.v,
                                                         (short)0, c0, false, false);
            c1 = __builtin_amdgcn_wmma_f32_16x16x32_bf16(false, a.v, false, b1.v,
                                                         (short)0, c1, false, false);
        }
    }

    // Store (VGPR i: row mtile*16 + halfsel*8 + i, col tile base + l16)
    float* Cp = Out + (size_t)(mtile * 16 + halfsel * 8) * Ncols + (nt0 * 16 + l16);
    #pragma unroll
    for (int i = 0; i < 8; ++i) {
        Cp[(size_t)i * Ncols]      = c0[i];
        Cp[(size_t)i * Ncols + 16] = c1[i];
    }
}

// ---------------------------------------------------------------------------
// Head: pooled[128] -> tanh(lin1) -> lin2 -> out[10]
// ---------------------------------------------------------------------------
__global__ void k_head(const float* __restrict__ pooled,
                       const float* __restrict__ W1, const float* __restrict__ b1,
                       const float* __restrict__ W2, const float* __restrict__ b2,
                       float* __restrict__ out) {
    __shared__ float o1[64];
    int t = threadIdx.x;
    if (t < 64) {
        float s = b1[t];
        for (int k = 0; k < 128; ++k) s += pooled[k] * W1[k * 64 + t];
        o1[t] = tanhf(s);
    }
    __syncthreads();
    if (t < 10) {
        float s = b2[t];
        for (int j = 0; j < 64; ++j) s += o1[j] * W2[j * 10 + t];
        out[t] = s;
    }
}

// ---------------------------------------------------------------------------
// Host orchestration
// ---------------------------------------------------------------------------
static inline unsigned gsz(size_t n, unsigned b) {
    size_t g = (n + b - 1) / b;
    if (g > 1048576) g = 1048576;
    return (unsigned)g;
}

extern "C" void kernel_launch(void* const* d_in, const int* in_sizes, int n_in,
                              void* d_out, int out_size, void* d_ws, size_t ws_size,
                              hipStream_t stream) {
    static const int CFG[N_LAYERS][3] = {
        {128,256,2},{256,512,3},{512,256,3},{256,128,2},
        {128,256,2},{256,512,3},{512,256,3},{256,128,2},
        {128,256,2},{256,512,3},{512,256,3},{256,128,2},
        {128,256,2},{256,512,3},{512,256,3},{256,128,2}};

    const float* x  = (const float*)d_in[0];
    const int*   ei = (const int*)d_in[1];
    const int N = in_sizes[0] / 128;
    const int E = in_sizes[1] / 2;
    const int* e_src = ei;
    const int* e_dst = ei + E;

    // Input index map (dict order): x, edge_index, batch, conv_Ws(40), conv_bs(16),
    // gn_params(12*3), lin1_W, lin1_b, lin2_W, lin2_b
    const float* Wf[N_LAYERS][3];
    size_t Woff[N_LAYERS][3];
    int widx = 3;
    size_t woff = 0;
    for (int i = 0; i < N_LAYERS; ++i) {
        for (int k = 0; k < CFG[i][2]; ++k) {
            Wf[i][k] = (const float*)d_in[widx++];
            Woff[i][k] = woff;
            woff += (size_t)CFG[i][0] * CFG[i][1];
        }
    }
    const int biasBase = widx;                // 43
    const int gnBase   = biasBase + N_LAYERS; // 59
    const int linBase  = gnBase + 12 * 3;     // 95
    const float* lin1W = (const float*)d_in[linBase + 0];
    const float* lin1b = (const float*)d_in[linBase + 1];
    const float* lin2W = (const float*)d_in[linBase + 2];
    const float* lin2b = (const float*)d_in[linBase + 3];

    // Workspace layout
    const size_t SZ = (size_t)N * 512;           // max activation elements
    float* bufA = (float*)d_ws;                  // T0 / current h (f32)
    float* bufB = bufA + SZ;                     // T1 (f32)
    float* bufC = bufB + SZ;                     // T2 (f32), then reused as GEMM output
    unsigned short* bf0 = (unsigned short*)(bufC + SZ);  // bf16 T0
    unsigned short* bf1 = bf0 + SZ;                      // bf16 T1
    unsigned short* bf2 = bf1 + SZ;                      // bf16 T2
    unsigned short* wbf = bf2 + SZ;                      // bf16 transposed weights
    float* deg   = (float*)(wbf + woff);
    float* dis   = deg + N;
    float* nrm   = dis + N;
    float* meanb = nrm + E;
    float* varb  = meanb + 512;

    // --- One-time prep: bf16/transposed weights ---
    for (int i = 0; i < N_LAYERS; ++i) {
        int cin = CFG[i][0], cout = CFG[i][1];
        for (int k = 0; k < CFG[i][2]; ++k) {
            size_t tot = (size_t)cin * cout;
            k_w_bf16_t<<<gsz(tot, 256), 256, 0, stream>>>(Wf[i][k], wbf + Woff[i][k], cin, cout);
        }
    }

    // --- Graph normalization coefficients ---
    k_zero_f32<<<gsz(N, 256), 256, 0, stream>>>(deg, (size_t)N);
    k_degree<<<gsz(E, 256), 256, 0, stream>>>(e_src, e_dst, deg, E);
    k_dis<<<gsz(N, 256), 256, 0, stream>>>(deg, dis, N);
    k_edge_norm<<<gsz(E, 256), 256, 0, stream>>>(e_src, e_dst, dis, nrm, E);

    // --- h = x ---
    k_copy_f32<<<gsz((size_t)N * 128, 256), 256, 0, stream>>>(x, bufA, (size_t)N * 128);

    int gn_i = 0;
    const int mtiles = (N + 15) / 16;
    for (int i = 0; i < N_LAYERS; ++i) {
        const int cin = CFG[i][0], cout = CFG[i][1], K = CFG[i][2];
        const size_t totIn = (size_t)N * cin;
        const size_t totOut = (size_t)N * cout;
        const float* bias = (const float*)d_in[biasBase + i];

        // T0 = h  -> bf16 staging (feeds both the GEMM and the T1 gather)
        k_f32_to_bf16<<<gsz(totIn, 256), 256, 0, stream>>>(bufA, bf0, totIn);

        // T1 = prop(T0): gather from bf16 copy, accumulate f32
        k_zero_f32<<<gsz(totIn, 256), 256, 0, stream>>>(bufB, totIn);
        k_spmm_scatter<<<E, 128, 0, stream>>>(e_src, e_dst, nrm, bf0, bufB, cin, E, 1.0f);
        k_f32_to_bf16<<<gsz(totIn, 256), 256, 0, stream>>>(bufB, bf1, totIn);

        if (K > 2) {
            // T2 = 2*prop(T1) - T0: seed with -T0, scatter 2*norm from bf16 T1
            k_negcopy_f32<<<gsz(totIn, 256), 256, 0, stream>>>(bufA, bufC, totIn);
            k_spmm_scatter<<<E, 128, 0, stream>>>(e_src, e_dst, nrm, bf1, bufC, cin, E, 2.0f);
            k_f32_to_bf16<<<gsz(totIn, 256), 256, 0, stream>>>(bufC, bf2, totIn);
        }

        // Fused GEMM: bufC = bias + sum_k T_k @ W_k  (accumulator stays in VGPRs)
        dim3 gemm_grid((mtiles + 7) / 8, cout / 32);
        if (K > 2) {
            k_wmma_gemm_fused<3><<<gemm_grid, 256, 0, stream>>>(
                bf0, bf1, bf2,
                wbf + Woff[i][0], wbf + Woff[i][1], wbf + Woff[i][2],
                bias, bufC, N, cin, cout);
        } else {
            k_wmma_gemm_fused<2><<<gemm_grid, 256, 0, stream>>>(
                bf0, bf1, bf1,
                wbf + Woff[i][0], wbf + Woff[i][1], wbf + Woff[i][1],
                bias, bufC, N, cin, cout);
        }

        if (i % 4 == 3) {
            // residual to original input + relu (cout == 128 here)
            k_resid_relu<<<gsz(totOut, 256), 256, 0, stream>>>(bufC, x, bufA, totOut);
        } else {
            const float* gw = (const float*)d_in[gnBase + gn_i * 3 + 0];
            const float* gb = (const float*)d_in[gnBase + gn_i * 3 + 1];
            const float* ms = (const float*)d_in[gnBase + gn_i * 3 + 2];
            ++gn_i;
            k_col_stats<<<cout, 256, 0, stream>>>(bufC, ms, meanb, varb, N, cout);
            k_gn_leaky<<<gsz(totOut, 256), 256, 0, stream>>>(bufC, bufA, meanb, varb, gw, gb, ms,
                                                             totOut, cout);
        }
    }

    // Global mean pool (128 cols) + MLP head
    k_col_mean<<<128, 256, 0, stream>>>(bufA, meanb, N, 128);
    k_head<<<1, 64, 0, stream>>>(meanb, lin1W, lin1b, lin2W, lin2b, (float*)d_out);
}
